// SeqWinAttn_32899449487849
// MI455X (gfx1250) — matrix operand order repaired and checked by reading, not compiled, not run on hardware
//
#include <hip/hip_runtime.h>

typedef __attribute__((ext_vector_type(8)))  float   v8f;
typedef __attribute__((ext_vector_type(4)))  __bf16  v4bf;
typedef __attribute__((ext_vector_type(8)))  __bf16  v8bf;
typedef __attribute__((ext_vector_type(16))) __bf16  v16bf;

#define DF    512
#define NH    8
#define DH    64
#define WIN   64
#define NWIN  64
#define BATCH 8
#define NTOK  4096

// LDS strides (in bf16 elements), padded for bank-conflict-free 16B fragment loads
#define XS  520   // x tile row stride (64 rows x 512 cols + 8 pad)
#define HS  72    // per-head 64x64 row stride (+8 pad)
#define HSZ (64 * HS)

static __device__ __forceinline__ v16bf ld_frag(const __bf16* p0, const __bf16* p1) {
  v8bf lo = *(const v8bf*)p0;
  v8bf hi = *(const v8bf*)p1;
  v16bf r;
#pragma unroll
  for (int i = 0; i < 8; ++i) { r[i] = lo[i]; r[i + 8] = hi[i]; }
  return r;
}

static __device__ __forceinline__ v8f zero8() {
  v8f z = {0.f, 0.f, 0.f, 0.f, 0.f, 0.f, 0.f, 0.f};
  return z;
}

// ---------------------------------------------------------------------------
// Kernel 0: one-shot fp32 -> bf16 conversion of both weight matrices (float4).
// ---------------------------------------------------------------------------
__global__ __launch_bounds__(256) void cvt_weights(const float* __restrict__ wqkv,
                                                   const float* __restrict__ projw,
                                                   __bf16* __restrict__ dq,
                                                   __bf16* __restrict__ dp) {
  int idx = blockIdx.x * 256 + threadIdx.x;
  const int NQ4 = 3 * DF * DF / 4;   // 196608
  const int NP4 = DF * DF / 4;       // 65536
  if (idx < NQ4) {
    float4 v = ((const float4*)wqkv)[idx];
    v4bf t = {(__bf16)v.x, (__bf16)v.y, (__bf16)v.z, (__bf16)v.w};
    ((v4bf*)dq)[idx] = t;
  } else if (idx < NQ4 + NP4) {
    int j = idx - NQ4;
    float4 v = ((const float4*)projw)[j];
    v4bf t = {(__bf16)v.x, (__bf16)v.y, (__bf16)v.z, (__bf16)v.w};
    ((v4bf*)dp)[j] = t;
  }
}

// ---------------------------------------------------------------------------
// Kernel 1: fused per-window QKV + attention.
// One block = one (batch, window). 256 threads = 8 waves. Wave w == head w
// in the attention stage. Dynamic LDS: 288 KB.
//   [0      , 73728)  x tile bf16 (64 x XS)  -> reused as P (8 heads x 64 x HS)
//   [73728  ,147456)  Q  (8 heads x 64 x HS), row-major (token, dh)
//   [147456 ,221184)  K  (same layout)
//   [221184 ,294912)  V^T (8 heads x 64 x HS), row-major (dh, token)
// ---------------------------------------------------------------------------
__global__ __launch_bounds__(256) void win_attn_qkv(const float* __restrict__ x,
                                                    const __bf16* __restrict__ wqkv,
                                                    __bf16* __restrict__ attn_out) {
  extern __shared__ __align__(16) char smem[];
  __bf16* Xl = (__bf16*)smem;
  __bf16* Ql = (__bf16*)(smem + 73728);
  __bf16* Kl = (__bf16*)(smem + 147456);
  __bf16* Vt = (__bf16*)(smem + 221184);

  const int tid  = threadIdx.x;
  const int wave = tid >> 5;
  const int lane = tid & 31;
  const int half = lane >> 4;
  const int l16  = lane & 15;

  const int blk = blockIdx.x;
  const int b   = blk >> 6;   // / NWIN
  const int w   = blk & 63;
  const size_t rowbase = (size_t)b * NTOK + (size_t)w * WIN;

  // ---- Stage 1: load x tile, convert to bf16 in LDS (packed b64 stores) ---
  const float4* xg = (const float4*)(x + rowbase * DF);
#pragma unroll 1
  for (int i = tid; i < WIN * DF / 4; i += 256) {
    float4 v = xg[i];
    int row = i >> 7;            // 128 float4 per row
    int c4  = (i & 127) << 2;
    v4bf t = {(__bf16)v.x, (__bf16)v.y, (__bf16)v.z, (__bf16)v.w};
    *(v4bf*)(Xl + row * XS + c4) = t;
  }
  __syncthreads();

  // ---- Stage 2: QKV = X(64x512) * Wqkv^T(512x1536) ------------------------
  // Each wave owns 12 N-tiles (16 cols each) and computes all 4 M-tiles for
  // each, holding 4 accumulators so every B fragment is loaded once from
  // global and reused by 4 WMMAs (4x less L2 weight traffic).
#pragma unroll 1
  for (int i = 0; i < 12; ++i) {
    const int et = wave + i * 8;             // wave-uniform N-tile index
    const int e0 = et << 4;
    v8f acc[4] = {zero8(), zero8(), zero8(), zero8()};
    const __bf16* brow = wqkv + (size_t)(e0 + l16) * DF + half * 8;
#pragma unroll 4
    for (int d0 = 0; d0 < DF; d0 += 32) {
      v16bf bm = ld_frag(brow + d0, brow + d0 + 16);
#pragma unroll
      for (int mt = 0; mt < 4; ++mt) {
        const __bf16* ap = Xl + (mt * 16 + l16) * XS + half * 8 + d0;
        v16bf a = ld_frag(ap, ap + 16);
        acc[mt] = __builtin_amdgcn_wmma_f32_16x16x32_bf16(false, a, false, bm,
                                                          (short)0, acc[mt], false, false);
      }
    }
    // Scatter into per-head Q / K / V^T. s and h are wave-uniform: force them
    // scalar so this is an s_cbranch, not an EXEC-masked 3-way branch.
    const int e0u   = __builtin_amdgcn_readfirstlane(e0);
    const int s     = e0u >> 9;            // 0=q, 1=k, 2=v
    const int h     = (e0u >> 6) & 7;      // head
    const int d     = (e0u & 63) + l16;    // dh index for this lane
    if (s == 0) {
      __bf16* dst = Ql + h * HSZ + d;
#pragma unroll
      for (int mt = 0; mt < 4; ++mt)
#pragma unroll
        for (int r = 0; r < 8; ++r)
          dst[(mt * 16 + r + half * 8) * HS] = (__bf16)acc[mt][r];
    } else if (s == 1) {
      __bf16* dst = Kl + h * HSZ + d;
#pragma unroll
      for (int mt = 0; mt < 4; ++mt)
#pragma unroll
        for (int r = 0; r < 8; ++r)
          dst[(mt * 16 + r + half * 8) * HS] = (__bf16)acc[mt][r];
    } else {
      __bf16* dst = Vt + h * HSZ + d * HS; // transposed: row = dh, col = token
#pragma unroll
      for (int mt = 0; mt < 4; ++mt)
#pragma unroll
        for (int r = 0; r < 8; ++r)
          dst[mt * 16 + r + half * 8] = (__bf16)acc[mt][r];
    }
  }
  __syncthreads();

  // ---- Stage 3: per-head attention (wave == head) -------------------------
  const int head = wave;
  const __bf16* qh = Ql + head * HSZ;
  const __bf16* kh = Kl + head * HSZ;
  const __bf16* vh = Vt + head * HSZ;
  __bf16*       Ph = Xl + head * HSZ;   // reuse x-tile region for P
  const float kExpScale = 0.125f * 1.4426950408889634f;  // DH^-0.5 * log2(e)

#pragma unroll 1
  for (int mi = 0; mi < 4; ++mi) {
    // S row-block: 16 query rows x 64 keys
    v8f sAcc[4] = {zero8(), zero8(), zero8(), zero8()};
    const __bf16* arow = qh + (mi * 16 + l16) * HS + half * 8;
#pragma unroll
    for (int k0 = 0; k0 < DH; k0 += 32) {
      v16bf a = ld_frag(arow + k0, arow + k0 + 16);
#pragma unroll
      for (int ni = 0; ni < 4; ++ni) {
        const __bf16* bp = kh + (ni * 16 + l16) * HS + k0 + half * 8;
        v16bf bm = ld_frag(bp, bp + 16);
        sAcc[ni] = __builtin_amdgcn_wmma_f32_16x16x32_bf16(false, a, false, bm,
                                                           (short)0, sAcc[ni], false, false);
      }
    }
    // Row softmax. Row M lives across one 16-lane half-group (N = l16).
#pragma unroll
    for (int r = 0; r < 8; ++r) {
      float m = fmaxf(fmaxf(sAcc[0][r], sAcc[1][r]), fmaxf(sAcc[2][r], sAcc[3][r]));
#pragma unroll
      for (int off = 8; off >= 1; off >>= 1)
        m = fmaxf(m, __shfl_xor(m, off, 16));
      float sum = 0.f;
#pragma unroll
      for (int ni = 0; ni < 4; ++ni) {
        float p = __builtin_amdgcn_exp2f((sAcc[ni][r] - m) * kExpScale);
        sAcc[ni][r] = p;
        sum += p;
      }
#pragma unroll
      for (int off = 8; off >= 1; off >>= 1)
        sum += __shfl_xor(sum, off, 16);
      const float inv = __builtin_amdgcn_rcpf(sum);
#pragma unroll
      for (int ni = 0; ni < 4; ++ni)
        Ph[(mi * 16 + r + half * 8) * HS + ni * 16 + l16] = (__bf16)(sAcc[ni][r] * inv);
    }
    // Wave-local LDS RAW fence: P stores must land before A-fragment reloads.
    asm volatile("s_wait_dscnt 0" ::: "memory");

    // O row-block = P(16x64) * V(64x64); B operand reads contiguous rows of V^T
    const __bf16* prow = Ph + (mi * 16 + l16) * HS + half * 8;
#pragma unroll
    for (int ni = 0; ni < 4; ++ni) {
      v8f o = zero8();
#pragma unroll
      for (int k0 = 0; k0 < WIN; k0 += 32) {
        v16bf a = ld_frag(prow + k0, prow + k0 + 16);
        const __bf16* bp = vh + (ni * 16 + l16) * HS + k0 + half * 8;
        v16bf bm = ld_frag(bp, bp + 16);
        o = __builtin_amdgcn_wmma_f32_16x16x32_bf16(false, a, false, bm,
                                                    (short)0, o, false, false);
      }
      const int feat = head * DH + ni * 16 + l16;
#pragma unroll
      for (int r = 0; r < 8; ++r) {
        const size_t row = rowbase + (size_t)(mi * 16 + r + half * 8);
        attn_out[row * DF + feat] = (__bf16)o[r];
      }
    }
  }
}

// ---------------------------------------------------------------------------
// Kernel 2: out = attn(32768x512, bf16) @ proj_w^T(512x512, bf16) + bias (f32)
// Block tile 128x64; wave holds one 16-row strip and 4 N-tile accumulators,
// reusing its A fragment across N.
// ---------------------------------------------------------------------------
__global__ __launch_bounds__(256) void proj_gemm(const __bf16* __restrict__ a,
                                                 const __bf16* __restrict__ wp,
                                                 const float* __restrict__ bias,
                                                 float* __restrict__ out) {
  const int tid  = threadIdx.x;
  const int wave = tid >> 5;
  const int lane = tid & 31;
  const int half = lane >> 4;
  const int l16  = lane & 15;

  const int m0 = (blockIdx.x >> 3) * 128 + wave * 16;
  const int e0 = (blockIdx.x & 7) * 64;

  v8f acc[4] = {zero8(), zero8(), zero8(), zero8()};
  const __bf16* arow = a + (size_t)(m0 + l16) * DF + half * 8;
#pragma unroll 4
  for (int d0 = 0; d0 < DF; d0 += 32) {
    v16bf af = ld_frag(arow + d0, arow + d0 + 16);
#pragma unroll
    for (int nt = 0; nt < 4; ++nt) {
      const __bf16* bp = wp + (size_t)(e0 + nt * 16 + l16) * DF + d0 + half * 8;
      v16bf bm = ld_frag(bp, bp + 16);
      acc[nt] = __builtin_amdgcn_wmma_f32_16x16x32_bf16(false, af, false, bm,
                                                        (short)0, acc[nt], false, false);
    }
  }
#pragma unroll
  for (int nt = 0; nt < 4; ++nt) {
    const int e = e0 + nt * 16 + l16;
    const float bb = bias[e];
#pragma unroll
    for (int r = 0; r < 8; ++r) {
      const int row = m0 + r + half * 8;
      out[(size_t)row * DF + e] = acc[nt][r] + bb;
    }
  }
}

// ---------------------------------------------------------------------------
extern "C" void kernel_launch(void* const* d_in, const int* in_sizes, int n_in,
                              void* d_out, int out_size, void* d_ws, size_t ws_size,
                              hipStream_t stream) {
  (void)in_sizes; (void)n_in; (void)out_size; (void)ws_size;
  const float* x     = (const float*)d_in[0];
  // d_in[1] = z : unused by the reference
  const float* wqkv  = (const float*)d_in[2];
  const float* projw = (const float*)d_in[3];
  const float* projb = (const float*)d_in[4];
  float* out = (float*)d_out;

  char* ws = (char*)d_ws;
  __bf16* ws_wqkv  = (__bf16*)ws;                                            // 1.5 MB
  __bf16* ws_projw = (__bf16*)(ws + (size_t)3 * DF * DF * 2);                // 0.5 MB
  __bf16* ws_attn  = (__bf16*)(ws + (size_t)3 * DF * DF * 2 + (size_t)DF * DF * 2); // 32 MB

  // weight conversion: (3*512*512 + 512*512)/4 float4 elements
  cvt_weights<<<1024, 256, 0, stream>>>(wqkv, projw, ws_wqkv, ws_projw);

  // fused QKV + windowed attention: one block per (batch, window)
  win_attn_qkv<<<BATCH * NWIN, 256, 294912, stream>>>(x, ws_wqkv, ws_attn);

  // output projection + bias
  proj_gemm<<<(BATCH * NTOK / 128) * (DF / 64), 256, 0, stream>>>(ws_attn, ws_projw,
                                                                  projb, out);
}
